// GlmDSAAttention_62895501082723
// MI455X (gfx1250) — matrix operand order repaired and checked by reading, not compiled, not run on hardware
//
#include <hip/hip_runtime.h>
#include <hip/hip_bf16.h>

// ---------------------------------------------------------------------------
// Types / helpers for CDNA5 WMMA (gfx1250, wave32)
// ---------------------------------------------------------------------------
typedef __attribute__((ext_vector_type(16))) __bf16 v16bf;
typedef __attribute__((ext_vector_type(8)))  float  v8f;
typedef __attribute__((ext_vector_type(4)))  unsigned int v4u;

#define WMMA_BF16(a, b, c) \
    __builtin_amdgcn_wmma_f32_16x16x32_bf16(false, (a), false, (b), (short)0, (c), false, false)

__device__ __forceinline__ unsigned short f32_to_bf16(float f) {
    unsigned u = __builtin_bit_cast(unsigned, f);
    unsigned r = u + 0x7FFFu + ((u >> 16) & 1u);
    return (unsigned short)(r >> 16);
}

// Load a 16-element bf16 fragment from two 16-byte chunks.
__device__ __forceinline__ v16bf load_frag(const unsigned short* p0, const unsigned short* p1) {
    union { v4u q[2]; v16bf v; } u;
    u.q[0] = *(const v4u*)p0;
    u.q[1] = *(const v4u*)p1;
    return u.v;
}

__device__ __forceinline__ v8f zero8() {
    v8f z = {0.f, 0.f, 0.f, 0.f, 0.f, 0.f, 0.f, 0.f};
    return z;
}

#define C_S     2048
#define C_HID   2048
#define C_H     16
#define C_NOPE  128
#define C_ROPE  64
#define C_VDIM  128
#define C_QR    1536
#define C_KVR   512
#define C_IH    16
#define C_ID    128
#define C_TOPK  512
#define C_NEG   (-1.0e9f)
#define C_SCALE 0.072168783648703219f /* 1/sqrt(192) */

// ---------------------------------------------------------------------------
// Elementwise f32 -> bf16 convert
// ---------------------------------------------------------------------------
__global__ __launch_bounds__(256) void cvt_f32_to_bf16_k(const float* __restrict__ x,
                                                         unsigned short* __restrict__ y,
                                                         long long n) {
    long long i = (long long)blockIdx.x * 256 + threadIdx.x;
    if (i < n) y[i] = f32_to_bf16(x[i]);
}

// ---------------------------------------------------------------------------
// RoPE cos/sin tables
// ---------------------------------------------------------------------------
__global__ __launch_bounds__(64) void rope_tab_k(float* __restrict__ cosb, float* __restrict__ sinb) {
    int s = blockIdx.x, j = threadIdx.x;
    int jj = j & 31;
    float e   = (float)(2 * jj) / 64.f;
    float inv = powf(10000.f, -e);
    float f   = (float)s * inv;
    cosb[s * 64 + j] = cosf(f);
    sinb[s * 64 + j] = sinf(f);
}

// ---------------------------------------------------------------------------
// WMMA GEMM:  C[M,N] (f32) = A[M,K] (bf16 row-major) x Bt[N,K]^T (bf16 row-major)
// Block: 128 threads = 4 waves. Wave tile 32(M) x 64(N) -> block tile 128x64.
// Ping-pong double-buffered (unroll x2): two fixed fragment register sets
// alternate roles, so there are no loop-carried register copies and no
// WMMA->VALU hazard NOPs; loads for one buffer overlap WMMAs on the other.
// Requires K % 64 == 0 (all K here: 2048/1536/512).
// ---------------------------------------------------------------------------
__global__ __launch_bounds__(128) void wmma_gemm_nt_k(const unsigned short* __restrict__ A,
                                                      const unsigned short* __restrict__ Bt,
                                                      float* __restrict__ C,
                                                      int M, int N, int K) {
    int lane = threadIdx.x & 31;
    int wave = threadIdx.x >> 5;
    int m0 = blockIdx.y * 128 + wave * 32;
    int n0 = blockIdx.x * 64;
    if (m0 >= M) return;
    int half = lane >> 4;
    int lm   = lane & 15;
    int akb  = half ? 8 : 0;
    int bkb  = half ? 8 : 0;

    v8f acc[8];
#pragma unroll
    for (int t = 0; t < 8; ++t) acc[t] = zero8();

    const unsigned short* a0 = A + (size_t)(m0 + lm) * K + akb;
    const unsigned short* a1 = A + (size_t)(m0 + 16 + lm) * K + akb;
    const unsigned short* bp[4];
#pragma unroll
    for (int t = 0; t < 4; ++t) bp[t] = Bt + (size_t)(n0 + t * 16 + lm) * K + bkb;

    // buffer X <- chunk 0
    v16bf a0x = load_frag(a0, a0 + 16);
    v16bf a1x = load_frag(a1, a1 + 16);
    v16bf bx[4], by[4];
#pragma unroll
    for (int t = 0; t < 4; ++t) bx[t] = load_frag(bp[t], bp[t] + 16);

    for (int k0 = 0; k0 < K; k0 += 64) {
        int k1 = k0 + 32;
        int k2 = (k0 + 64 < K) ? (k0 + 64) : 0; // wraparound dummy on last iter

        // buffer Y <- chunk k1 (in flight while X computes)
        v16bf a0y = load_frag(a0 + k1, a0 + k1 + 16);
        v16bf a1y = load_frag(a1 + k1, a1 + k1 + 16);
#pragma unroll
        for (int t = 0; t < 4; ++t) by[t] = load_frag(bp[t] + k1, bp[t] + k1 + 16);

        if (k0 + 256 < K) { // pull future A rows toward the WGP (global_prefetch)
            __builtin_prefetch((const void*)(a0 + k0 + 256), 0, 0);
            __builtin_prefetch((const void*)(a1 + k0 + 256), 0, 0);
        }

        // compute chunk k0 from buffer X
#pragma unroll
        for (int t = 0; t < 4; ++t) {
            acc[t]     = WMMA_BF16(a0x, bx[t], acc[t]);
            acc[4 + t] = WMMA_BF16(a1x, bx[t], acc[4 + t]);
        }

        // buffer X <- chunk k2 (in flight while Y computes)
        a0x = load_frag(a0 + k2, a0 + k2 + 16);
        a1x = load_frag(a1 + k2, a1 + k2 + 16);
#pragma unroll
        for (int t = 0; t < 4; ++t) bx[t] = load_frag(bp[t] + k2, bp[t] + k2 + 16);

        // compute chunk k1 from buffer Y
#pragma unroll
        for (int t = 0; t < 4; ++t) {
            acc[t]     = WMMA_BF16(a0y, by[t], acc[t]);
            acc[4 + t] = WMMA_BF16(a1y, by[t], acc[4 + t]);
        }
    }

#pragma unroll
    for (int t = 0; t < 4; ++t) {
        int n = n0 + t * 16 + lm;
        if (n < N) {
#pragma unroll
            for (int r = 0; r < 8; ++r) {
                int ma = m0 + r + half * 8;
                int mb = ma + 16;
                C[(size_t)ma * N + n] = acc[t][r];
                C[(size_t)mb * N + n] = acc[4 + t][r];
            }
        }
    }
}

// ---------------------------------------------------------------------------
// Row-wise RMSNorm (f32 in, bf16 out)
// ---------------------------------------------------------------------------
__global__ __launch_bounds__(256) void rmsnorm_bf16_k(const float* __restrict__ x, int ldx,
                                                      const float* __restrict__ w,
                                                      unsigned short* __restrict__ y, int ldy, int D) {
    __shared__ float red[256];
    int row = blockIdx.x, t = threadIdx.x;
    const float* xr = x + (size_t)row * ldx;
    float ss = 0.f;
    for (int i = t; i < D; i += 256) { float v = xr[i]; ss += v * v; }
    red[t] = ss; __syncthreads();
    for (int o = 128; o > 0; o >>= 1) { if (t < o) red[t] += red[t + o]; __syncthreads(); }
    float inv = rsqrtf(red[0] / (float)D + 1e-6f);
    for (int i = t; i < D; i += 256)
        y[(size_t)row * ldy + i] = f32_to_bf16(xr[i] * inv * w[i]);
}

// ---------------------------------------------------------------------------
// k_pe rope
// ---------------------------------------------------------------------------
__global__ __launch_bounds__(64) void kpe_rope_k(const float* __restrict__ ckv,
                                                 const float* __restrict__ cosb,
                                                 const float* __restrict__ sinb,
                                                 unsigned short* __restrict__ kpe) {
    __shared__ float rp[64];
    int s = blockIdx.x, j = threadIdx.x;
    float v = ckv[(size_t)s * (C_KVR + C_ROPE) + C_KVR + j];
    rp[j] = v; __syncthreads();
    float rot = (j < 32) ? -rp[j + 32] : rp[j - 32];
    float o = v * cosb[s * 64 + j] + rot * sinb[s * 64 + j];
    kpe[(size_t)s * 64 + j] = f32_to_bf16(o);
}

// ---------------------------------------------------------------------------
// qh assemble
// ---------------------------------------------------------------------------
__global__ __launch_bounds__(192) void qh_assemble_k(const float* __restrict__ qraw,
                                                     const float* __restrict__ cosb,
                                                     const float* __restrict__ sinb,
                                                     unsigned short* __restrict__ qh) {
    __shared__ float rp[64];
    int s = blockIdx.x, h = blockIdx.y, j = threadIdx.x;
    size_t base = ((size_t)s * C_H + h) * 192;
    float v = qraw[base + j];
    if (j >= 128) rp[j - 128] = v;
    __syncthreads();
    float o;
    if (j < 128) {
        o = v;
    } else {
        int p = j - 128;
        float rot = (p < 32) ? -rp[p + 32] : rp[p - 32];
        o = rp[p] * cosb[s * 64 + p] + rot * sinb[s * 64 + p];
    }
    qh[base + j] = f32_to_bf16(o);
}

// ---------------------------------------------------------------------------
// kh / vT assemble
// ---------------------------------------------------------------------------
__global__ __launch_bounds__(192) void kh_v_assemble_k(const float* __restrict__ kvraw,
                                                       const unsigned short* __restrict__ kpe,
                                                       unsigned short* __restrict__ kh,
                                                       unsigned short* __restrict__ vT, int S) {
    int s = blockIdx.x, h = blockIdx.y, j = threadIdx.x;
    size_t kb = ((size_t)s * C_H + h) * 256;
    size_t ob = ((size_t)s * C_H + h) * 192;
    if (j < 128) {
        kh[ob + j] = f32_to_bf16(kvraw[kb + j]);
        vT[((size_t)h * C_VDIM + j) * (size_t)S + s] = f32_to_bf16(kvraw[kb + 128 + j]);
    } else {
        kh[ob + j] = kpe[(size_t)s * 64 + (j - 128)];
    }
}

// ---------------------------------------------------------------------------
// iq assemble
// ---------------------------------------------------------------------------
__global__ __launch_bounds__(128) void iq_assemble_k(const float* __restrict__ iqraw,
                                                     const float* __restrict__ cosb,
                                                     const float* __restrict__ sinb,
                                                     unsigned short* __restrict__ iq) {
    __shared__ float rp[64];
    int s = blockIdx.x, h = blockIdx.y, j = threadIdx.x;
    size_t base = ((size_t)s * C_IH + h) * C_ID;
    float v = iqraw[base + j];
    if (j < 64) rp[j] = v;
    __syncthreads();
    float o;
    if (j < 64) {
        float rot = (j < 32) ? -rp[j + 32] : rp[j - 32];
        o = rp[j] * cosb[s * 64 + j] + rot * sinb[s * 64 + j];
    } else {
        o = v;
    }
    iq[base + j] = f32_to_bf16(o);
}

// ---------------------------------------------------------------------------
// ik: layernorm + rope
// ---------------------------------------------------------------------------
__global__ __launch_bounds__(128) void ik_ln_rope_k(const float* __restrict__ x,
                                                    const float* __restrict__ w,
                                                    const float* __restrict__ b,
                                                    const float* __restrict__ cosb,
                                                    const float* __restrict__ sinb,
                                                    unsigned short* __restrict__ y) {
    __shared__ float ls[128], red[128];
    int s = blockIdx.x, j = threadIdx.x;
    float v = x[(size_t)s * C_ID + j];
    red[j] = v; __syncthreads();
    for (int o = 64; o > 0; o >>= 1) { if (j < o) red[j] += red[j + o]; __syncthreads(); }
    float mean = red[0] / 128.f; __syncthreads();
    float d = v - mean;
    red[j] = d * d; __syncthreads();
    for (int o = 64; o > 0; o >>= 1) { if (j < o) red[j] += red[j + o]; __syncthreads(); }
    float var = red[0] / 128.f;
    float ln = d * rsqrtf(var + 1e-6f) * w[j] + b[j];
    ls[j] = ln; __syncthreads();
    float o;
    if (j < 64) {
        float rot = (j < 32) ? -ls[j + 32] : ls[j - 32];
        o = ls[j] * cosb[s * 64 + j] + rot * sinb[s * 64 + j];
    } else {
        o = ln;
    }
    y[(size_t)s * C_ID + j] = f32_to_bf16(o);
}

// ---------------------------------------------------------------------------
// iw[s,h] = (hs[s,:] . wproj[h,:]) * IH^{-1/2}
// ---------------------------------------------------------------------------
__global__ __launch_bounds__(256) void iw_k(const float* __restrict__ hs,
                                            const float* __restrict__ wp,
                                            float* __restrict__ iw, int n) {
    int i = blockIdx.x * 256 + threadIdx.x;
    if (i >= n) return;
    int s = i >> 4, h = i & 15;
    const float* a = hs + (size_t)s * C_HID;
    const float* c = wp + (size_t)h * C_HID;
    float acc = 0.f;
    for (int d = 0; d < C_HID; ++d) acc += a[d] * c[d];
    iw[i] = acc * 0.25f;
}

// ---------------------------------------------------------------------------
// Indexer scores via WMMA. ik B-fragments head-invariant (hoisted); iq
// A-fragments ping-pong double-buffered across a x2-unrolled head loop.
// ---------------------------------------------------------------------------
__global__ __launch_bounds__(128) void idx_scores_k(const unsigned short* __restrict__ iq,
                                                    const unsigned short* __restrict__ ik,
                                                    const float* __restrict__ iw,
                                                    float* __restrict__ sc, int S) {
    int lane = threadIdx.x & 31;
    int wave = threadIdx.x >> 5;
    int q0 = blockIdx.y * 16;
    int k0 = blockIdx.x * 64 + wave * 16;
    int half = lane >> 4;
    int lm   = lane & 15;

    if (k0 > q0 + 15) { // fully above diagonal -> masked
#pragma unroll
        for (int r = 0; r < 8; ++r) {
            int m = q0 + r + half * 8;
            sc[(size_t)m * S + k0 + lm] = C_NEG;
        }
        return;
    }

    int akb = half ? 8 : 0;
    int bkb = half ? 8 : 0;

    // ik fragments: reused for every head
    v16bf bik[4];
    const unsigned short* bbase = ik + (size_t)(k0 + lm) * C_ID + bkb;
#pragma unroll
    for (int kk = 0; kk < 4; ++kk) bik[kk] = load_frag(bbase + kk * 32, bbase + kk * 32 + 16);

    const unsigned short* abase = iq + (size_t)(q0 + lm) * C_IH * C_ID + akb;
    v16bf ax[4], ay[4];
#pragma unroll
    for (int kk = 0; kk < 4; ++kk) ax[kk] = load_frag(abase + kk * 32, abase + kk * 32 + 16);

    float tot[8];
#pragma unroll
    for (int r = 0; r < 8; ++r) tot[r] = 0.f;

    for (int h = 0; h < C_IH; h += 2) {
        // buffer Y <- head h+1
        const unsigned short* apy = abase + (size_t)(h + 1) * C_ID;
#pragma unroll
        for (int kk = 0; kk < 4; ++kk) ay[kk] = load_frag(apy + kk * 32, apy + kk * 32 + 16);

        // compute head h from buffer X
        v8f c0 = zero8();
#pragma unroll
        for (int kk = 0; kk < 4; ++kk) c0 = WMMA_BF16(ax[kk], bik[kk], c0);
#pragma unroll
        for (int r = 0; r < 8; ++r) {
            int m = q0 + r + half * 8;
            float v = c0[r];
            v = v > 0.f ? v : 0.f;
            tot[r] += iw[m * C_IH + h] * v;
        }

        // buffer X <- head h+2 (wraparound dummy on last iteration)
        int hn = (h + 2 < C_IH) ? (h + 2) : 0;
        const unsigned short* apx = abase + (size_t)hn * C_ID;
#pragma unroll
        for (int kk = 0; kk < 4; ++kk) ax[kk] = load_frag(apx + kk * 32, apx + kk * 32 + 16);

        // compute head h+1 from buffer Y
        v8f c1 = zero8();
#pragma unroll
        for (int kk = 0; kk < 4; ++kk) c1 = WMMA_BF16(ay[kk], bik[kk], c1);
#pragma unroll
        for (int r = 0; r < 8; ++r) {
            int m = q0 + r + half * 8;
            float v = c1[r];
            v = v > 0.f ? v : 0.f;
            tot[r] += iw[m * C_IH + h + 1] * v;
        }
    }

#pragma unroll
    for (int r = 0; r < 8; ++r) {
        int m = q0 + r + half * 8;
        int k = k0 + lm;
        sc[(size_t)m * S + k] = (k <= m) ? tot[r] : C_NEG;
    }
}

// ---------------------------------------------------------------------------
// Per-row top-k threshold by value bisection
// ---------------------------------------------------------------------------
__global__ __launch_bounds__(256) void topk_thresh_k(const float* __restrict__ sc,
                                                     float* __restrict__ thr, int S) {
    __shared__ float smx[256], smn[256];
    __shared__ int   icnt[256];
    int q = blockIdx.x, t = threadIdx.x;
    if (q + 1 <= C_TOPK) { if (t == 0) thr[q] = -3.0e38f; return; }
    const float* row = sc + (size_t)q * S;
    float mx = -3.0e38f, mn = 3.0e38f;
    for (int k = t; k <= q; k += 256) { float v = row[k]; mx = fmaxf(mx, v); mn = fminf(mn, v); }
    smx[t] = mx; smn[t] = mn; __syncthreads();
    for (int o = 128; o > 0; o >>= 1) {
        if (t < o) { smx[t] = fmaxf(smx[t], smx[t + o]); smn[t] = fminf(smn[t], smn[t + o]); }
        __syncthreads();
    }
    float hi = smx[0], lo = smn[0];
    for (int it = 0; it < 28; ++it) {
        float mid = 0.5f * (lo + hi);
        int c = 0;
        for (int k = t; k <= q; k += 256) c += (row[k] >= mid) ? 1 : 0;
        icnt[t] = c; __syncthreads();
        for (int o = 128; o > 0; o >>= 1) { if (t < o) icnt[t] += icnt[t + o]; __syncthreads(); }
        int total = icnt[0];
        __syncthreads();
        if (total >= C_TOPK) lo = mid; else hi = mid;
    }
    if (t == 0) thr[q] = lo;
}

// ---------------------------------------------------------------------------
// Masked flash attention with WMMA (bf16 inputs, f32 online softmax).
// Block 128 thr = 4 waves; wave = one 16-query tile of one head.
// Inner loops fully unrolled (SSA; no loop-carried fragment copies).
// ---------------------------------------------------------------------------
__global__ __launch_bounds__(128) void attn_k(const unsigned short* __restrict__ qh,
                                              const unsigned short* __restrict__ kh,
                                              const unsigned short* __restrict__ vT,
                                              const float* __restrict__ isc,
                                              const float* __restrict__ thr,
                                              float* __restrict__ out, int S) {
    __shared__ unsigned short pbuf[4][16][32];
    int lane = threadIdx.x & 31;
    int wave = threadIdx.x >> 5;
    int h  = blockIdx.y;
    int q0 = (blockIdx.x * 4 + wave) * 16;
    int half = lane >> 4;
    int lm   = lane & 15;
    int akb = half ? 8 : 0;
    int bkb = half ? 8 : 0;

    // Preload q fragments: 6 chunks of 32 along the 192-dim feature axis.
    v16bf aq[6];
    const unsigned short* qp = qh + ((size_t)(q0 + lm) * C_H + h) * 192;
#pragma unroll
    for (int c = 0; c < 6; ++c)
        aq[c] = load_frag(qp + c * 32 + akb, qp + c * 32 + akb + 16);

    v8f o[8];
#pragma unroll
    for (int t = 0; t < 8; ++t) o[t] = zero8();
    float mrow[8], lrow[8], th[8];
#pragma unroll
    for (int r = 0; r < 8; ++r) {
        mrow[r] = -3.0e38f; lrow[r] = 0.f;
        th[r] = thr[q0 + r + half * 8];
    }

    int kend = q0 + 16; // causal upper bound (exclusive)
    for (int kc = 0; kc < kend; kc += 32) {
        // --- score tiles S0 (keys kc..kc+15), S1 (keys kc+16..kc+31)
        const unsigned short* kb0 = kh + ((size_t)(kc + lm) * C_H + h) * 192 + bkb;
        const unsigned short* kb1 = kh + ((size_t)(kc + 16 + lm) * C_H + h) * 192 + bkb;
        v8f s0 = zero8(), s1 = zero8();
        v16bf b0 = load_frag(kb0, kb0 + 16);
        v16bf b1 = load_frag(kb1, kb1 + 16);
#pragma unroll
        for (int c = 0; c < 6; ++c) {
            v16bf b0n = b0, b1n = b1;
            if (c < 5) {
                b0n = load_frag(kb0 + (c + 1) * 32, kb0 + (c + 1) * 32 + 16);
                b1n = load_frag(kb1 + (c + 1) * 32, kb1 + (c + 1) * 32 + 16);
            }
            s0 = WMMA_BF16(aq[c], b0, s0);
            s1 = WMMA_BF16(aq[c], b1, s1);
            b0 = b0n; b1 = b1n;
        }

        // --- mask, online-softmax update, write P (bf16) to LDS
        float corr[8];
#pragma unroll
        for (int r = 0; r < 8; ++r) {
            int m   = q0 + r + half * 8;
            int k0g = kc + lm;
            int k1g = kc + 16 + lm;
            float v0 = s0[r] * C_SCALE;
            float v1 = s1[r] * C_SCALE;
            if (!((k0g <= m) && (isc[(size_t)m * S + k0g] >= th[r]))) v0 = C_NEG;
            if (!((k1g <= m) && (isc[(size_t)m * S + k1g] >= th[r]))) v1 = C_NEG;
            float mx = fmaxf(v0, v1);
            mx = fmaxf(mx, __shfl_xor(mx, 1, 32));
            mx = fmaxf(mx, __shfl_xor(mx, 2, 32));
            mx = fmaxf(mx, __shfl_xor(mx, 4, 32));
            mx = fmaxf(mx, __shfl_xor(mx, 8, 32));
            float mnew = fmaxf(mrow[r], mx);
            float p0 = __expf(v0 - mnew);
            float p1 = __expf(v1 - mnew);
            float rs = p0 + p1;
            rs += __shfl_xor(rs, 1, 32);
            rs += __shfl_xor(rs, 2, 32);
            rs += __shfl_xor(rs, 4, 32);
            rs += __shfl_xor(rs, 8, 32);
            float cc = __expf(mrow[r] - mnew);
            lrow[r] = lrow[r] * cc + rs;
            mrow[r] = mnew;
            corr[r] = cc;
            pbuf[wave][r + half * 8][lm]      = f32_to_bf16(p0);
            pbuf[wave][r + half * 8][lm + 16] = f32_to_bf16(p1);
        }

        // rescale accumulators before adding new P*V
#pragma unroll
        for (int t = 0; t < 8; ++t)
#pragma unroll
            for (int r = 0; r < 8; ++r) o[t][r] *= corr[r];

        // CDNA5 split LDS counter: ensure P-tile writes complete before reload.
        asm volatile("s_wait_dscnt 0x0" ::: "memory");

        // reload P as a 16x32 A-fragment from LDS
        const unsigned short* lp = &pbuf[wave][lm][half ? 8 : 0];
        v16bf aP = load_frag(lp, lp + 16);

        // P (16x32) x V (32x16) for 8 output tiles; V fragments pipelined.
        const unsigned short* vbase =
            vT + ((size_t)h * C_VDIM + lm) * (size_t)S + kc + bkb;
        v16bf bV = load_frag(vbase, vbase + 16);
#pragma unroll
        for (int t = 0; t < 8; ++t) {
            v16bf bVn = bV;
            if (t < 7) {
                const unsigned short* vp = vbase + (size_t)(t + 1) * 16 * S;
                bVn = load_frag(vp, vp + 16);
            }
            o[t] = WMMA_BF16(aP, bV, o[t]);
            bV = bVn;
        }
    }

    // epilogue: normalize and store [s][h*128 + d]
#pragma unroll
    for (int t = 0; t < 8; ++t)
#pragma unroll
        for (int r = 0; r < 8; ++r) {
            int m = q0 + r + half * 8;
            out[(size_t)m * (C_H * C_VDIM) + h * C_VDIM + t * 16 + lm] = o[t][r] / lrow[r];
        }
}

// ---------------------------------------------------------------------------
// Host-side orchestration
// ---------------------------------------------------------------------------
extern "C" void kernel_launch(void* const* d_in, const int* in_sizes, int n_in,
                              void* d_out, int out_size, void* d_ws, size_t ws_size,
                              hipStream_t stream) {
    (void)in_sizes; (void)n_in; (void)out_size; (void)ws_size;
    const int S = C_S, HID = C_HID, H = C_H, QR = C_QR, KVR = C_KVR, IH = C_IH, ID = C_ID;

    const float* hs          = (const float*)d_in[0];
    const float* q_a_w       = (const float*)d_in[1];
    const float* q_a_ln_w    = (const float*)d_in[2];
    const float* q_b_w       = (const float*)d_in[3];
    const float* kv_a_w      = (const float*)d_in[4];
    const float* kv_a_ln_w   = (const float*)d_in[5];
    const float* kv_b_w      = (const float*)d_in[6];
    const float* o_w         = (const float*)d_in[7];
    const float* idx_wq_b_w  = (const float*)d_in[8];
    const float* idx_wk_w    = (const float*)d_in[9];
    const float* idx_kn_w    = (const float*)d_in[10];
    const float* idx_kn_b    = (const float*)d_in[11];
    const float* idx_wproj_w = (const float*)d_in[12];

    char* ws = (char*)d_ws;
    size_t off = 0;
    auto alloc = [&](size_t bytes) -> void* {
        void* p = ws + off;
        off = (off + bytes + 255) & ~(size_t)255;
        return p;
    };

    unsigned short* hs_b   = (unsigned short*)alloc((size_t)S * HID * 2);
    unsigned short* qaw_b  = (unsigned short*)alloc((size_t)QR * HID * 2);
    unsigned short* qbw_b  = (unsigned short*)alloc((size_t)H * 192 * QR * 2);
    unsigned short* kvaw_b = (unsigned short*)alloc((size_t)(KVR + 64) * HID * 2);
    unsigned short* kvbw_b = (unsigned short*)alloc((size_t)H * 256 * KVR * 2);
    unsigned short* ow_b   = (unsigned short*)alloc((size_t)HID * H * C_VDIM * 2);
    unsigned short* iqw_b  = (unsigned short*)alloc((size_t)IH * ID * QR * 2);
    unsigned short* ikw_b  = (unsigned short*)alloc((size_t)ID * HID * 2);
    float* cosb            = (float*)alloc((size_t)S * 64 * 4);
    float* sinb            = (float*)alloc((size_t)S * 64 * 4);
    float* qr_raw          = (float*)alloc((size_t)S * QR * 4);
    unsigned short* qr_b   = (unsigned short*)alloc((size_t)S * QR * 2);
    float* q_raw           = (float*)alloc((size_t)S * H * 192 * 4);
    float* ckv_raw         = (float*)alloc((size_t)S * (KVR + 64) * 4);
    unsigned short* ckv_b  = (unsigned short*)alloc((size_t)S * KVR * 2);
    unsigned short* kpe_b  = (unsigned short*)alloc((size_t)S * 64 * 2);
    float* kv_raw          = (float*)alloc((size_t)S * H * 256 * 4);
    unsigned short* qh_b   = (unsigned short*)alloc((size_t)S * H * 192 * 2);
    unsigned short* kh_b   = (unsigned short*)alloc((size_t)S * H * 192 * 2);
    unsigned short* vT_b   = (unsigned short*)alloc((size_t)H * C_VDIM * S * 2);
    float* iq_raw          = (float*)alloc((size_t)S * IH * ID * 4);
    unsigned short* iq_b   = (unsigned short*)alloc((size_t)S * IH * ID * 2);
    float* ik_raw          = (float*)alloc((size_t)S * ID * 4);
    unsigned short* ik_b   = (unsigned short*)alloc((size_t)S * ID * 2);
    float* iw_f            = (float*)alloc((size_t)S * IH * 4);
    float* isc             = (float*)alloc((size_t)S * S * 4);
    float* thr             = (float*)alloc((size_t)S * 4);
    float* attn_f          = (float*)alloc((size_t)S * H * C_VDIM * 4);
    unsigned short* attn_b = (unsigned short*)alloc((size_t)S * H * C_VDIM * 2);

    auto cvt = [&](const float* src, unsigned short* dst, size_t n) {
        cvt_f32_to_bf16_k<<<dim3((unsigned)((n + 255) / 256)), dim3(256), 0, stream>>>(
            src, dst, (long long)n);
    };
    auto gemm = [&](const unsigned short* A, const unsigned short* Bt, float* C,
                    int M, int N, int K) {
        wmma_gemm_nt_k<<<dim3((unsigned)(N / 64), (unsigned)(M / 128)), dim3(128), 0, stream>>>(
            A, Bt, C, M, N, K);
    };

    // weight / activation conversions + rope tables
    cvt(hs, hs_b, (size_t)S * HID);
    cvt(q_a_w, qaw_b, (size_t)QR * HID);
    cvt(q_b_w, qbw_b, (size_t)H * 192 * QR);
    cvt(kv_a_w, kvaw_b, (size_t)(KVR + 64) * HID);
    cvt(kv_b_w, kvbw_b, (size_t)H * 256 * KVR);
    cvt(o_w, ow_b, (size_t)HID * H * C_VDIM);
    cvt(idx_wq_b_w, iqw_b, (size_t)IH * ID * QR);
    cvt(idx_wk_w, ikw_b, (size_t)ID * HID);
    rope_tab_k<<<dim3(S), dim3(64), 0, stream>>>(cosb, sinb);

    // q path
    gemm(hs_b, qaw_b, qr_raw, S, QR, HID);
    rmsnorm_bf16_k<<<dim3(S), dim3(256), 0, stream>>>(qr_raw, QR, q_a_ln_w, qr_b, QR, QR);
    gemm(qr_b, qbw_b, q_raw, S, H * 192, QR);

    // kv path
    gemm(hs_b, kvaw_b, ckv_raw, S, KVR + 64, HID);
    rmsnorm_bf16_k<<<dim3(S), dim3(256), 0, stream>>>(ckv_raw, KVR + 64, kv_a_ln_w, ckv_b, KVR, KVR);
    kpe_rope_k<<<dim3(S), dim3(64), 0, stream>>>(ckv_raw, cosb, sinb, kpe_b);
    gemm(ckv_b, kvbw_b, kv_raw, S, H * 256, KVR);

    // assemble attention operands
    qh_assemble_k<<<dim3(S, H), dim3(192), 0, stream>>>(q_raw, cosb, sinb, qh_b);
    kh_v_assemble_k<<<dim3(S, H), dim3(192), 0, stream>>>(kv_raw, kpe_b, kh_b, vT_b, S);

    // indexer path
    gemm(qr_b, iqw_b, iq_raw, S, IH * ID, QR);
    iq_assemble_k<<<dim3(S, IH), dim3(128), 0, stream>>>(iq_raw, cosb, sinb, iq_b);
    gemm(hs_b, ikw_b, ik_raw, S, ID, HID);
    ik_ln_rope_k<<<dim3(S), dim3(128), 0, stream>>>(ik_raw, idx_kn_w, idx_kn_b, cosb, sinb, ik_b);
    iw_k<<<dim3((S * IH) / 256), dim3(256), 0, stream>>>(hs, idx_wproj_w, iw_f, S * IH);
    idx_scores_k<<<dim3(S / 64, S / 16), dim3(128), 0, stream>>>(iq_b, ik_b, iw_f, isc, S);
    topk_thresh_k<<<dim3(S), dim3(256), 0, stream>>>(isc, thr, S);

    // sparse-masked flash attention
    attn_k<<<dim3(S / 64, H), dim3(128), 0, stream>>>(qh_b, kh_b, vT_b, isc, thr, attn_f, S);

    // output projection
    cvt(attn_f, attn_b, (size_t)S * H * C_VDIM);
    gemm(attn_b, ow_b, (float*)d_out, S, HID, H * C_VDIM);
}
